// Attention_69707319214091
// MI455X (gfx1250) — matrix operand order repaired and checked by reading, not compiled, not run on hardware
//
#include <hip/hip_runtime.h>

#define BB 32
#define QQ 1024
#define KK 1024
#define DD 1024

typedef __attribute__((ext_vector_type(16))) __bf16 bf16x16;
typedef __attribute__((ext_vector_type(8)))  float  f32x8;
typedef __attribute__((ext_vector_type(4)))  unsigned int u32x4;
typedef __attribute__((ext_vector_type(8)))  int i32x8;
typedef __attribute__((ext_vector_type(4)))  int i32x4;

union FragBF { bf16x16 f; uint4 u[2]; };

__device__ __forceinline__ unsigned short f32_to_bf16(float x) {
  unsigned int u = __float_as_uint(x);
  u += 0x7FFFu + ((u >> 16) & 1u);   // round to nearest even
  return (unsigned short)(u >> 16);
}

// ---- fragment loads from GLOBAL (row-major, ld in elements) ----
__device__ __forceinline__ void load_fragA(FragBF& fr, const unsigned short* base,
                                           int ld, int row0, int c0, int lane) {
  const int r = row0 + (lane & 15);
  const int c = c0 + ((lane >> 4) << 3);
  const uint4* p = (const uint4*)(base + (size_t)r * ld + c);
  fr.u[0] = p[0];
  fr.u[1] = p[2];
}
__device__ __forceinline__ void load_fragB(FragBF& fr, const unsigned short* base,
                                           int ld, int row0, int c0, int lane) {
  const int r = row0 + (lane & 15);
  const int c = c0 + ((lane >> 4) << 3);
  const uint4* p = (const uint4*)(base + (size_t)r * ld + c);
  fr.u[0] = p[0];
  fr.u[1] = p[2];
}

// ---- fragment loads from an LDS tile laid out [rows][32] bf16 ----
// Take the shared array by reference so the address space stays LDS (ds_load).
__device__ __forceinline__ void lds_fragA(FragBF& fr,
                                          const unsigned short (&tile)[128 * 32],
                                          int row0, int lane) {
  const int r = row0 + (lane & 15);
  const int c = (lane >> 4) << 3;
  fr.u[0] = *(const uint4*)&tile[r * 32 + c];
  fr.u[1] = *(const uint4*)&tile[r * 32 + c + 16];
}
__device__ __forceinline__ void lds_fragB(FragBF& fr,
                                          const unsigned short (&tile)[128 * 32],
                                          int row0, int lane) {
  const int r = row0 + (lane & 15);
  const int c = (lane >> 4) << 3;
  fr.u[0] = *(const uint4*)&tile[r * 32 + c];
  fr.u[1] = *(const uint4*)&tile[r * 32 + c + 16];
}

__device__ __forceinline__ f32x8 wmma_bf16(const FragBF& a, const FragBF& b, f32x8 c) {
  return __builtin_amdgcn_wmma_f32_16x16x32_bf16(false, a.f, false, b.f,
                                                 (short)0, c, false, false);
}

// ---- Tensor Data Mover: 2D tile (tile_rows x tile_cols bf16) -> LDS ----
__device__ __forceinline__ void tdm_load_tile_2d(const unsigned short* lds_dst,
                                                 const unsigned short* gsrc,
                                                 int row_pitch,   // elements
                                                 int tile_cols,   // elements (X)
                                                 int tile_rows) { // rows (Y)
  const unsigned long long ga = (unsigned long long)(const void*)gsrc;
  const unsigned lds_off = (unsigned)(unsigned long long)(const void*)lds_dst;
  u32x4 g0 = { 1u,                                    // count=1, user mode
               lds_off,                               // lds_addr
               (unsigned)ga,                          // global_addr[31:0]
               (unsigned)((ga >> 32) & 0x01FFFFFFu)   // global_addr[56:32]
                 | (2u << 30) };                      // type=2 (image)
  i32x8 g1 = { 1 << 16,                               // mask=0, data_size=1 (2B)
               (int)((tile_cols & 0xFFFF) << 16),     // tensor_dim0[15:0]
               (int)((tile_rows & 0xFFFF) << 16),     // dim0 hi=0 | tensor_dim1[15:0]
               (int)((tile_cols & 0xFFFF) << 16),     // dim1 hi=0 | tile_dim0
               (int)(tile_rows & 0xFFFF),             // tile_dim1 | tile_dim2=0
               row_pitch,                             // tensor_dim0_stride[31:0]
               0, 0 };                                // stride hi / dim1_stride
  i32x4 g2 = { 1, 1, 0, 0 };                          // tensor_dim2=1, dim3=1
  i32x4 g3 = { 0, 1 << 16, 0, 0 };                    // dim3_stride=0, tensor_dim4=1
#if defined(__clang_major__) && __clang_major__ >= 23
  i32x8 z8 = { 0, 0, 0, 0, 0, 0, 0, 0 };
  __builtin_amdgcn_tensor_load_to_lds(g0, g1, g2, g3, z8, 0);
#else
  __builtin_amdgcn_tensor_load_to_lds(g0, g1, g2, g3, 0);
#endif
}

// ---------------------------------------------------------------- converts
__global__ void __launch_bounds__(256)
f32_to_bf16_kernel(const float* __restrict__ src, unsigned short* __restrict__ dst,
                   long n4) {
  long stride = (long)gridDim.x * blockDim.x;
  for (long i = (long)blockIdx.x * blockDim.x + threadIdx.x; i < n4; i += stride) {
    float4 f = ((const float4*)src)[i];
    unsigned int lo  = ((unsigned)f32_to_bf16(f.y) << 16) | f32_to_bf16(f.x);
    unsigned int hi2 = ((unsigned)f32_to_bf16(f.w) << 16) | f32_to_bf16(f.z);
    ((uint2*)dst)[i] = make_uint2(lo, hi2);
  }
}

// ------------------------------------------------- context transpose (bf16)
__global__ void __launch_bounds__(256)
transpose_bf16_kernel(const unsigned short* __restrict__ src,
                      unsigned short* __restrict__ dst) {
  __shared__ unsigned short t[32][33];
  const int b  = blockIdx.z;
  const int k0 = blockIdx.y * 32;
  const int d0 = blockIdx.x * 32;
  const unsigned short* s = src + (size_t)b * KK * DD;
  unsigned short*       o = dst + (size_t)b * DD * KK;
  const int tx = threadIdx.x, ty = threadIdx.y;
  #pragma unroll
  for (int i = 0; i < 32; i += 8)
    t[ty + i][tx] = s[(size_t)(k0 + ty + i) * DD + d0 + tx];
  __syncthreads();
  #pragma unroll
  for (int i = 0; i < 32; i += 8)
    o[(size_t)(d0 + ty + i) * KK + k0 + tx] = t[tx][ty + i];
}

// ------------------------------------------------------ TDM-staged GEMM
// C[b] = A0[b]*B0[b]^T (+ A1[b]*B1[b]^T) (+bias) (tanh?).  A [M,Dk], B [N,Dk]
// row-major. 128x128 block tile, 8 waves x (64x32). A/B k-slabs staged into
// LDS by the Tensor Data Mover, double buffered, synced with TENSORcnt+barrier.
__device__ __forceinline__ void gemm_tdm_pass(
    f32x8 (&acc)[4][2],
    const unsigned short* Abase, const unsigned short* Bbase,
    int lda, int ldb, int Dk, int gm0, int gn0, int m0l, int n0l,
    int wave_u, int lane,
    unsigned short (&shA)[2][128 * 32],
    unsigned short (&shB)[2][128 * 32]) {
  const int nsteps = Dk >> 5;
  const unsigned short* Atile = Abase + (size_t)gm0 * lda;
  const unsigned short* Btile = Bbase + (size_t)gn0 * ldb;

  if (wave_u == 0) {
    tdm_load_tile_2d(&shA[0][0], Atile, lda, 32, 128);
    tdm_load_tile_2d(&shB[0][0], Btile, ldb, 32, 128);
    if (nsteps > 1) {
      tdm_load_tile_2d(&shA[1][0], Atile + 32, lda, 32, 128);
      tdm_load_tile_2d(&shB[1][0], Btile + 32, ldb, 32, 128);
    }
  }
  for (int i = 0; i < nsteps; i++) {
    if (wave_u == 0) {
      if (i + 1 < nsteps) __builtin_amdgcn_s_wait_tensorcnt(2);
      else                __builtin_amdgcn_s_wait_tensorcnt(0);
    }
    __syncthreads();                       // buf[i&1] visible to all waves
    const int buf = i & 1;
    FragBF a[4], b[2];
    #pragma unroll
    for (int x = 0; x < 4; x++) lds_fragA(a[x], shA[buf], m0l + x * 16, lane);
    #pragma unroll
    for (int y = 0; y < 2; y++) lds_fragB(b[y], shB[buf], n0l + y * 16, lane);
    #pragma unroll
    for (int x = 0; x < 4; x++)
      #pragma unroll
      for (int y = 0; y < 2; y++) acc[x][y] = wmma_bf16(a[x], b[y], acc[x][y]);
    __syncthreads();                       // all waves done reading buf[i&1]
    if (wave_u == 0 && i + 2 < nsteps) {
      const int d0 = (i + 2) * 32;
      tdm_load_tile_2d(&shA[buf][0], Atile + d0, lda, 32, 128);
      tdm_load_tile_2d(&shB[buf][0], Btile + d0, ldb, 32, 128);
    }
  }
}

__global__ void __launch_bounds__(256)
gemm_abt_kernel(const unsigned short* __restrict__ A0, long sA0,
                const unsigned short* __restrict__ B0, long sB0,
                const unsigned short* __restrict__ A1, long sA1,
                const unsigned short* __restrict__ B1, long sB1,
                int lda, int ldb, int Dk, int M, int N,
                const float* __restrict__ bias,
                float* __restrict__ Cf, unsigned short* __restrict__ Cbf,
                int do_tanh) {
  __shared__ unsigned short shA[2][128 * 32];
  __shared__ unsigned short shB[2][128 * 32];

  const int lane   = threadIdx.x & 31;
  const int wave   = threadIdx.x >> 5;
  const int wave_u = __builtin_amdgcn_readfirstlane(threadIdx.x >> 5);
  const int batch  = blockIdx.z;
  const int gm0 = blockIdx.y * 128;
  const int gn0 = blockIdx.x * 128;
  const int m0l = (wave & 1) * 64;
  const int n0l = (wave >> 1) * 32;

  const f32x8 z = {0.f, 0.f, 0.f, 0.f, 0.f, 0.f, 0.f, 0.f};
  f32x8 acc[4][2];
  #pragma unroll
  for (int i = 0; i < 4; i++)
    #pragma unroll
    for (int j = 0; j < 2; j++) acc[i][j] = z;

  gemm_tdm_pass(acc, A0 + (size_t)batch * sA0, B0 + (size_t)batch * sB0,
                lda, ldb, Dk, gm0, gn0, m0l, n0l, wave_u, lane, shA, shB);
  if (A1) {
    __syncthreads();
    gemm_tdm_pass(acc, A1 + (size_t)batch * sA1, B1 + (size_t)batch * sB1,
                  lda, ldb, Dk, gm0, gn0, m0l, n0l, wave_u, lane, shA, shB);
  }

  const int hi = lane >> 4;
  const int nl = lane & 15;
  #pragma unroll
  for (int i = 0; i < 4; i++)
    #pragma unroll
    for (int j = 0; j < 2; j++)
      #pragma unroll
      for (int v = 0; v < 8; v++) {
        const int m = gm0 + m0l + i * 16 + v + hi * 8;
        const int n = gn0 + n0l + j * 16 + nl;
        float val = acc[i][j][v];
        if (bias) val += bias[n];
        if (do_tanh) val = tanhf(val);
        const size_t idx = ((size_t)batch * M + m) * N + n;
        if (Cf)  Cf[idx]  = val;
        if (Cbf) Cbf[idx] = f32_to_bf16(val);
      }
}

// ---------------------------------------- fused scores + mask + softmax
__global__ void __launch_bounds__(256)
scores_softmax_kernel(const unsigned short* __restrict__ Qbf,  // [B,Q,D] bf16
                      const unsigned short* __restrict__ Ctx,  // [B,K,D] bf16
                      const float* __restrict__ qmask,         // [B,Q,D], col 0
                      const float* __restrict__ cmask,         // [B,K,D], col 0
                      float* __restrict__ Wf,                  // [B,Q,K] fp32 out
                      unsigned short* __restrict__ Wbf) {      // [B,Q,K] bf16
  __shared__ float red[16][8];
  __shared__ float rowred[16];

  const int lane  = threadIdx.x & 31;
  const int wave  = threadIdx.x >> 5;
  const int batch = blockIdx.y;
  const int q0    = blockIdx.x * 16;
  const int n0w   = wave * 128;
  const int hi = lane >> 4, nl = lane & 15;

  const unsigned short* qb = Qbf + (size_t)batch * QQ * DD;
  const unsigned short* cb = Ctx + (size_t)batch * KK * DD;

  const f32x8 zv = {0.f, 0.f, 0.f, 0.f, 0.f, 0.f, 0.f, 0.f};
  f32x8 acc[8];
  #pragma unroll
  for (int t = 0; t < 8; t++) acc[t] = zv;

  for (int d0 = 0; d0 < DD; d0 += 32) {
    FragBF a;
    load_fragA(a, qb, DD, q0, d0, lane);
    #pragma unroll
    for (int t = 0; t < 8; t++) {
      FragBF b;
      load_fragB(b, cb, DD, n0w + t * 16, d0, lane);
      acc[t] = wmma_bf16(a, b, acc[t]);
    }
  }

  float qm[8], cm[8];
  #pragma unroll
  for (int v = 0; v < 8; v++)
    qm[v] = qmask[((size_t)batch * QQ + q0 + v + hi * 8) * DD];
  #pragma unroll
  for (int t = 0; t < 8; t++)
    cm[t] = cmask[((size_t)batch * KK + n0w + t * 16 + nl) * DD];
  #pragma unroll
  for (int t = 0; t < 8; t++)
    #pragma unroll
    for (int v = 0; v < 8; v++)
      acc[t][v] += (1.0f - qm[v] * cm[t]) * -10000.0f;

  float pm[8];
  #pragma unroll
  for (int v = 0; v < 8; v++) {
    pm[v] = -3.0e38f;
    #pragma unroll
    for (int t = 0; t < 8; t++) pm[v] = fmaxf(pm[v], acc[t][v]);
  }
  #pragma unroll
  for (int s = 1; s < 16; s <<= 1)
    #pragma unroll
    for (int v = 0; v < 8; v++) pm[v] = fmaxf(pm[v], __shfl_xor(pm[v], s, 32));
  if (nl == 0) {
    #pragma unroll
    for (int v = 0; v < 8; v++) red[v + hi * 8][wave] = pm[v];
  }
  __syncthreads();
  if (threadIdx.x < 16) {
    float m = red[threadIdx.x][0];
    #pragma unroll
    for (int w = 1; w < 8; w++) m = fmaxf(m, red[threadIdx.x][w]);
    rowred[threadIdx.x] = m;
  }
  __syncthreads();
  float rmax[8];
  #pragma unroll
  for (int v = 0; v < 8; v++) rmax[v] = rowred[v + hi * 8];

  float ps[8];
  #pragma unroll
  for (int v = 0; v < 8; v++) ps[v] = 0.f;
  #pragma unroll
  for (int t = 0; t < 8; t++)
    #pragma unroll
    for (int v = 0; v < 8; v++) {
      float e = __expf(acc[t][v] - rmax[v]);
      acc[t][v] = e;
      ps[v] += e;
    }
  #pragma unroll
  for (int s = 1; s < 16; s <<= 1)
    #pragma unroll
    for (int v = 0; v < 8; v++) ps[v] += __shfl_xor(ps[v], s, 32);
  if (nl == 0) {
    #pragma unroll
    for (int v = 0; v < 8; v++) red[v + hi * 8][wave] = ps[v];
  }
  __syncthreads();
  if (threadIdx.x < 16) {
    float m = red[threadIdx.x][0];
    #pragma unroll
    for (int w = 1; w < 8; w++) m += red[threadIdx.x][w];
    rowred[threadIdx.x] = m;
  }
  __syncthreads();
  float inv[8];
  #pragma unroll
  for (int v = 0; v < 8; v++) inv[v] = 1.0f / rowred[v + hi * 8];

  #pragma unroll
  for (int t = 0; t < 8; t++)
    #pragma unroll
    for (int v = 0; v < 8; v++) {
      const float wv = acc[t][v] * inv[v];
      const int m = q0 + v + hi * 8;
      const int n = n0w + t * 16 + nl;
      const size_t idx = ((size_t)batch * QQ + m) * KK + n;
      Wf[idx]  = wv;
      Wbf[idx] = f32_to_bf16(wv);
    }
}

// ----------------------------------------------------------------- launch
extern "C" void kernel_launch(void* const* d_in, const int* in_sizes, int n_in,
                              void* d_out, int out_size, void* d_ws, size_t ws_size,
                              hipStream_t stream) {
  (void)in_sizes; (void)n_in; (void)out_size; (void)ws_size;

  const float* query = (const float*)d_in[0];
  const float* ctx   = (const float*)d_in[1];
  const float* qmask = (const float*)d_in[2];
  const float* cmask = (const float*)d_in[3];
  const float* W_in  = (const float*)d_in[4];
  const float* b_in  = (const float*)d_in[5];
  const float* W_out = (const float*)d_in[6];
  const float* b_out = (const float*)d_in[7];

  float* out_f = (float*)d_out;                  // [B,Q,D]
  float* wts_f = out_f + (size_t)BB * QQ * DD;   // [B,Q,K]

  unsigned char* ws = (unsigned char*)d_ws;
  size_t o = 0;
  unsigned short* qry_bf  = (unsigned short*)(ws + o); o += (size_t)BB * QQ * DD * 2;
  unsigned short* ctx_bf  = (unsigned short*)(ws + o); o += (size_t)BB * KK * DD * 2;
  unsigned short* ctxT_bf = (unsigned short*)(ws + o); o += (size_t)BB * DD * KK * 2;
  unsigned short* win_bf  = (unsigned short*)(ws + o); o += (size_t)DD * DD * 2;
  unsigned short* wout_bf = (unsigned short*)(ws + o); o += (size_t)DD * 2 * DD * 2;
  unsigned short* q_bf    = (unsigned short*)(ws + o); o += (size_t)BB * QQ * DD * 2;
  unsigned short* wts_bf  = (unsigned short*)(ws + o); o += (size_t)BB * QQ * KK * 2;
  unsigned short* mix_bf  = (unsigned short*)(ws + o); o += (size_t)BB * QQ * DD * 2;

  f32_to_bf16_kernel<<<2048, 256, 0, stream>>>(query, qry_bf, (long)BB * QQ * DD / 4);
  f32_to_bf16_kernel<<<2048, 256, 0, stream>>>(ctx,   ctx_bf, (long)BB * KK * DD / 4);
  f32_to_bf16_kernel<<<64,   256, 0, stream>>>(W_in,  win_bf, (long)DD * DD / 4);
  f32_to_bf16_kernel<<<128,  256, 0, stream>>>(W_out, wout_bf, (long)DD * 2 * DD / 4);

  transpose_bf16_kernel<<<dim3(DD / 32, KK / 32, BB), dim3(32, 8), 0, stream>>>(
      ctx_bf, ctxT_bf);

  // q = query @ W_in^T + b_in -> q_bf
  gemm_abt_kernel<<<dim3(DD / 128, QQ / 128, BB), 256, 0, stream>>>(
      qry_bf, (long)QQ * DD, win_bf, 0L, nullptr, 0L, nullptr, 0L,
      DD, DD, DD, QQ, DD, b_in, nullptr, q_bf, 0);

  // weights = softmax(q @ ctx^T + mask) -> wts_f (output #2) and wts_bf
  scores_softmax_kernel<<<dim3(QQ / 16, BB), 256, 0, stream>>>(
      q_bf, ctx_bf, qmask, cmask, wts_f, wts_bf);

  // mix = weights @ context   (A = weights [Q,K], B = ctx^T [D,K])
  gemm_abt_kernel<<<dim3(DD / 128, QQ / 128, BB), 256, 0, stream>>>(
      wts_bf, (long)QQ * KK, ctxT_bf, (long)DD * KK, nullptr, 0L, nullptr, 0L,
      KK, KK, KK, QQ, DD, nullptr, nullptr, mix_bf, 0);

  // out = tanh(mix @ W_out[:, :D]^T + q @ W_out[:, D:]^T + b_out) -> d_out
  gemm_abt_kernel<<<dim3(DD / 128, QQ / 128, BB), 256, 0, stream>>>(
      mix_bf, (long)QQ * DD, wout_bf, 0L, q_bf, (long)QQ * DD, wout_bf + DD, 0L,
      DD, 2 * DD, DD, QQ, DD, b_out, out_f, nullptr, 1);
}
